// GMAN_84765474554251
// MI455X (gfx1250) — matrix operand order repaired and checked by reading, not compiled, not run on hardware
//
#include <hip/hip_runtime.h>
#include <hip/hip_bf16.h>

// ---------------------------------------------------------------------------
// GMAN block forward for MI455X (gfx1250, wave32, WMMA).
// All GEMM-like work (linears, ctx convs, inception causal convs, spatial
// attention) runs through v_wmma_f32_16x16x32_f16 with f32->f16 conversion
// fused into operand loads and f32 accumulation. Each GEMM wave computes a
// 16x64 tile (4 WMMAs share one A fragment). Conv weights are repacked once
// per call to [tap][O][I] and V is transposed per (b,t,head) so every WMMA
// B fragment is a contiguous vector load. Boundary masking is arithmetic
// (clamped row + x0 multiply) so loads stay clause-batched, never predicated.
// ---------------------------------------------------------------------------

#define Bb 8
#define Tt 12
#define Nn 400
#define Kh 8
#define Dd 128
#define Mtok (Bb*Tt*Nn)          /* 38400 tokens; 16 | Nn so tiles stay in (b,t) */
#define NPAD 416

typedef __attribute__((ext_vector_type(16))) _Float16 v16h;
typedef __attribute__((ext_vector_type(8)))  float    v8f;

// ---- WMMA operand loaders (f32 memory -> f16 regs, CDNA5 wave32 layouts) ----
// A 16x32 f16: row r = lane&15, h = lane>>4; elems 0..7 -> K=h*8+j,
// elems 8..15 -> K=16+h*8+j  (ISA 7.12.2).
__device__ __forceinline__ v16h load_a_f32(const float* arow, int h) {
  v16h a;
#pragma unroll
  for (int j = 0; j < 8; ++j) a[j]   = (_Float16)arow[h*8 + j];
#pragma unroll
  for (int j = 0; j < 8; ++j) a[8+j] = (_Float16)arow[16 + h*8 + j];
  return a;
}
// Masked variant: loads execute unconditionally (caller clamps the row to a
// valid address); msk is 0.f or 1.f and zeroes the fragment arithmetically.
__device__ __forceinline__ v16h load_a_f32_m(const float* arow, int h, float msk) {
  v16h a;
#pragma unroll
  for (int j = 0; j < 8; ++j) a[j]   = (_Float16)(arow[h*8 + j] * msk);
#pragma unroll
  for (int j = 0; j < 8; ++j) a[8+j] = (_Float16)(arow[16 + h*8 + j] * msk);
  return a;
}
// B 32x16 f16: col c = lane&15, h = lane>>4; elems 0..7 -> K=h*8+j,
// elems 8..15 -> K=16+h*8+j.
__device__ __forceinline__ v16h load_b_f32(const float* wrow, int h) {
  v16h b;
#pragma unroll
  for (int j = 0; j < 8; ++j) b[j]   = (_Float16)wrow[h*8 + j];
#pragma unroll
  for (int j = 0; j < 8; ++j) b[8+j] = (_Float16)wrow[16 + h*8 + j];
  return b;
}
__device__ __forceinline__ v8f v8f_zero() {
  v8f c;
#pragma unroll
  for (int g = 0; g < 8; ++g) c[g] = 0.f;
  return c;
}
#define WMMA_F16(a, b, c) \
  __builtin_amdgcn_wmma_f32_16x16x32_f16(false, (a), false, (b), (short)0, (c), false, false)

// ---------------------------------------------------------------------------
// Linear: out[m,n] = epi( sum_k A[m,k]*W[n,k] + bias[n] ),  W:[NOUT,KIN].
// One wave computes a 16x64 tile: 4 accumulators share each A fragment.
// ---------------------------------------------------------------------------
template<int KIN, int NOUT, bool RELU, bool HAS_RESID>
__global__ __launch_bounds__(128) void gemm_lin_t(
    const float* __restrict__ A, const float* __restrict__ W,
    const float* __restrict__ bias, const float* __restrict__ resid,
    float resid_scale, float* __restrict__ out)
{
  constexpr int NT = NOUT / 64;
  int wave = blockIdx.x * 4 + (threadIdx.x >> 5);
  int mt = wave / NT, nt = wave - mt * NT;
  if (mt * 16 >= Mtok) return;
  int lane = threadIdx.x & 31, r = lane & 15, h = lane >> 4;
  int m0 = mt * 16, n0 = nt * 64;
  const float* arow = A + (size_t)(m0 + r) * KIN;
  const float* wrow = W + (size_t)(n0 + r) * KIN;
  v8f c0 = v8f_zero(), c1 = v8f_zero(), c2 = v8f_zero(), c3 = v8f_zero();
  for (int k0 = 0; k0 < KIN; k0 += 32) {
    v16h a  = load_a_f32(arow + k0, h);
    v16h b0 = load_b_f32(wrow + k0, h);
    v16h b1 = load_b_f32(wrow + 16 * KIN + k0, h);
    v16h b2 = load_b_f32(wrow + 32 * KIN + k0, h);
    v16h b3 = load_b_f32(wrow + 48 * KIN + k0, h);
    c0 = WMMA_F16(a, b0, c0);
    c1 = WMMA_F16(a, b1, c1);
    c2 = WMMA_F16(a, b2, c2);
    c3 = WMMA_F16(a, b3, c3);
  }
  v8f cc[4] = {c0, c1, c2, c3};
#pragma unroll
  for (int i = 0; i < 4; ++i) {
    int col = n0 + 16 * i + r;
    float bv = bias[col];
#pragma unroll
    for (int g = 0; g < 8; ++g) {
      size_t o = (size_t)(m0 + g + 8 * h) * NOUT + col;
      float v = cc[i][g] + bv;
      if (RELU) v = v > 0.f ? v : 0.f;
      if (HAS_RESID) v += resid_scale * resid[o];
      out[o] = v;
    }
  }
}

// ---------------------------------------------------------------------------
// Conv weight repack: Win[O][I][KK] -> Wout[tap][O][I]  (contiguous K loads).
// ---------------------------------------------------------------------------
__global__ __launch_bounds__(256) void repack_conv_kernel(
    const float* __restrict__ Win, float* __restrict__ Wout, int KK)
{
  int i = blockIdx.x * blockDim.x + threadIdx.x;
  if (i >= Dd * Dd * KK) return;
  int tap = i / (Dd * Dd);
  int rem = i - tap * Dd * Dd;
  int o = rem >> 7, ii = rem & 127;
  Wout[i] = Win[(size_t)(o * Dd + ii) * KK + tap];
}

// ---------------------------------------------------------------------------
// Context conv over node axis: y[b,t,n,o] = sum_{dn=-1..1} sum_i
//   A[b,t,n+dn,i] * Wp[dn+1][o][i] + bias[o].   Zero-pad n; 16x64 tiles.
// ---------------------------------------------------------------------------
__global__ __launch_bounds__(128) void gemm_ctxconv(
    const float* __restrict__ A, const float* __restrict__ Wp,
    const float* __restrict__ bias, float* __restrict__ out)
{
  int wave = blockIdx.x * 4 + (threadIdx.x >> 5);
  int mt = wave >> 1, nt = wave & 1;
  if (mt * 16 >= Mtok) return;
  int lane = threadIdx.x & 31, r = lane & 15, h = lane >> 4;
  int m0 = mt * 16, n0 = nt * 64;
  int nrow = (m0 % Nn) + r;                       // node index of this lane's row
  v8f c[4] = {v8f_zero(), v8f_zero(), v8f_zero(), v8f_zero()};
#pragma unroll
  for (int tap = 0; tap < 3; ++tap) {
    int dn = tap - 1;
    bool ok = (unsigned)(nrow + dn) < (unsigned)Nn;
    float msk = ok ? 1.f : 0.f;
    // clamp to a valid row when masked; loads execute unconditionally and
    // the fragment is zeroed by the x msk multiply (keeps loads clause-batched)
    const float* arow = A + (ok ? (size_t)(m0 + r + dn) * Dd : (size_t)0);
    const float* wrow = Wp + (size_t)tap * Dd * Dd + (size_t)(n0 + r) * Dd;
    for (int k0 = 0; k0 < Dd; k0 += 32) {
      v16h a = load_a_f32_m(arow + k0, h, msk);
#pragma unroll
      for (int i = 0; i < 4; ++i) {
        v16h b = load_b_f32(wrow + 16 * i * Dd + k0, h);
        c[i] = WMMA_F16(a, b, c[i]);
      }
    }
  }
#pragma unroll
  for (int i = 0; i < 4; ++i) {
    int col = n0 + 16 * i + r;
    float bv = bias[col];
#pragma unroll
    for (int g = 0; g < 8; ++g)
      out[(size_t)(m0 + g + 8 * h) * Dd + col] = c[i][g] + bv;
  }
}

// ---------------------------------------------------------------------------
// Causal conv over time (inception branch): y = relu(conv_k(A) + bias),
// then merged += mW[i] * y.  Wp:[KK][O][I]; zero-pad t<0; tiles share t.
// ---------------------------------------------------------------------------
template<int KK>
__global__ __launch_bounds__(128) void gemm_causalconv(
    const float* __restrict__ A, const float* __restrict__ Wp,
    const float* __restrict__ bias, const float* __restrict__ accum_scale,
    float* __restrict__ out)
{
  int wave = blockIdx.x * 4 + (threadIdx.x >> 5);
  int mt = wave >> 1, nt = wave & 1;
  if (mt * 16 >= Mtok) return;
  int lane = threadIdx.x & 31, r = lane & 15, h = lane >> 4;
  int m0 = mt * 16, n0 = nt * 64;
  int t0 = (m0 / Nn) % Tt;
  v8f c[4] = {v8f_zero(), v8f_zero(), v8f_zero(), v8f_zero()};
#pragma unroll
  for (int tap = 0; tap < KK; ++tap) {
    int dt = tap - (KK - 1);
    bool ok = (t0 + dt) >= 0;
    float msk = ok ? 1.f : 0.f;
    const float* arow = A + (ok ? (size_t)(m0 + r + dt * Nn) * Dd : (size_t)0);
    const float* wrow = Wp + (size_t)tap * Dd * Dd + (size_t)(n0 + r) * Dd;
    for (int k0 = 0; k0 < Dd; k0 += 32) {
      v16h a = load_a_f32_m(arow + k0, h, msk);
#pragma unroll
      for (int i = 0; i < 4; ++i) {
        v16h b = load_b_f32(wrow + 16 * i * Dd + k0, h);
        c[i] = WMMA_F16(a, b, c[i]);
      }
    }
  }
  float asc = accum_scale[0];
#pragma unroll
  for (int i = 0; i < 4; ++i) {
    int col = n0 + 16 * i + r;
    float bv = bias[col];
#pragma unroll
    for (int g = 0; g < 8; ++g) {
      size_t o = (size_t)(m0 + g + 8 * h) * Dd + col;
      float v = c[i][g] + bv;
      v = v > 0.f ? v : 0.f;
      out[o] = out[o] + asc * v;
    }
  }
}

// ---------------------------------------------------------------------------
// LayerNorm over D=128, one wave per token; optional fused post-add.
// ---------------------------------------------------------------------------
template<bool POST_ADD>
__global__ __launch_bounds__(128) void ln_t(
    const float* __restrict__ src, const float* __restrict__ gamma,
    const float* __restrict__ beta, const float* __restrict__ post_add,
    float* __restrict__ out)
{
  int token = blockIdx.x * 4 + (threadIdx.x >> 5);
  if (token >= Mtok) return;
  int lane = threadIdx.x & 31;
  const float* s = src + (size_t)token * Dd;
  float v[4], sum = 0.f;
#pragma unroll
  for (int i = 0; i < 4; ++i) { v[i] = s[lane + 32 * i]; sum += v[i]; }
#pragma unroll
  for (int off = 16; off > 0; off >>= 1) sum += __shfl_xor(sum, off, 32);
  float mean = sum * (1.f / Dd);
  float var = 0.f;
#pragma unroll
  for (int i = 0; i < 4; ++i) { float d = v[i] - mean; var += d * d; }
#pragma unroll
  for (int off = 16; off > 0; off >>= 1) var += __shfl_xor(var, off, 32);
  float inv = rsqrtf(var * (1.f / Dd) + 1e-5f);
#pragma unroll
  for (int i = 0; i < 4; ++i) {
    int cidx = lane + 32 * i;
    float r = (v[i] - mean) * inv * gamma[cidx] + beta[cidx];
    if (POST_ADD) r += post_add[(size_t)token * Dd + cidx];
    out[(size_t)token * Dd + cidx] = r;
  }
}

// ---- small elementwise kernels ----
__global__ __launch_bounds__(256) void add_te_kernel(
    float* __restrict__ q, const float* __restrict__ te,
    const float* __restrict__ W2, const float* __restrict__ b2)
{
  int i = blockIdx.x * blockDim.x + threadIdx.x;
  if (i >= Mtok * Dd) return;
  int m = i >> 7, o = i & 127;
  q[i] += te[(size_t)m * 2] * W2[o * 2] + te[(size_t)m * 2 + 1] * W2[o * 2 + 1] + b2[o];
}
__global__ __launch_bounds__(256) void merge_init_kernel(
    float* __restrict__ merged, const float* __restrict__ y0,
    const float* __restrict__ mW)
{
  int i = blockIdx.x * blockDim.x + threadIdx.x;
  if (i >= Mtok * Dd) return;
  merged[i] = mW[0] * y0[i];
}
__global__ __launch_bounds__(256) void concat_kernel(
    const float* __restrict__ h1, const float* __restrict__ h2,
    float* __restrict__ cat)
{
  int i = blockIdx.x * blockDim.x + threadIdx.x;
  if (i >= Mtok * Dd) return;
  int m = i >> 7, o = i & 127;
  cat[(size_t)m * 256 + o]       = h1[i];
  cat[(size_t)m * 256 + 128 + o] = h2[i];
}
__global__ __launch_bounds__(256) void gate_kernel(
    const float* __restrict__ gl, const float* __restrict__ h1,
    const float* __restrict__ h2, float* __restrict__ ht)
{
  int i = blockIdx.x * blockDim.x + threadIdx.x;
  if (i >= Mtok * Dd) return;
  float g = 1.f / (1.f + __expf(-gl[i]));
  ht[i] = g * h1[i] + (1.f - g) * h2[i];
}
// bias[n,m] = alpha * <ne[n], ne[m]> + beta * lap[n,m]
__global__ __launch_bounds__(256) void graph_bias_kernel(
    const float* __restrict__ ne, const float* __restrict__ lap,
    const float* __restrict__ alpha, const float* __restrict__ beta,
    float* __restrict__ biasNM)
{
  int i = blockIdx.x * blockDim.x + threadIdx.x;
  if (i >= Nn * Nn) return;
  int n = i / Nn, m = i - n * Nn;
  float d = 0.f;
#pragma unroll
  for (int e = 0; e < 16; ++e) d += ne[n * 16 + e] * ne[m * 16 + e];
  biasNM[i] = alpha[0] * d + beta[0] * lap[i];
}
// V[(bt*400+m)*128 + kk*16+dd] -> Vt[(bt*8+kk)*16*NPAD + dd*NPAD + m] (pad->0)
__global__ __launch_bounds__(256) void vtrans_kernel(
    const float* __restrict__ V, float* __restrict__ Vt)
{
  int i = blockIdx.x * blockDim.x + threadIdx.x;
  if (i >= Bb * Tt * Kh * 16 * NPAD) return;
  int m = i % NPAD; int t1 = i / NPAD;
  int dd = t1 & 15; int hk = t1 >> 4;
  int kk = hk & 7,  bt = hk >> 3;
  float v = 0.f;
  if (m < Nn) v = V[((size_t)bt * Nn + m) * Dd + kk * 16 + dd];
  Vt[i] = v;
}

// ---------------------------------------------------------------------------
// Temporal attention over T=12 (per b,n,head): tiny -> VALU, 1 thread per row.
// ---------------------------------------------------------------------------
__global__ __launch_bounds__(256) void tattn_kernel(
    const float* __restrict__ Q, const float* __restrict__ K,
    const float* __restrict__ V, float* __restrict__ out)
{
  int tid = blockIdx.x * blockDim.x + threadIdx.x;
  if (tid >= Bb * Nn * Kh * Tt) return;
  int qt = tid % Tt; int t1 = tid / Tt;
  int kk = t1 % Kh;  int t2 = t1 / Kh;
  int n  = t2 % Nn;  int b  = t2 / Nn;
  int ho = kk * 16;
  const float* qrow = Q + (((size_t)b * Tt + qt) * Nn + n) * Dd + ho;
  float qv[16];
#pragma unroll
  for (int d = 0; d < 16; ++d) qv[d] = qrow[d];
  float s[Tt], mx = -1e30f;
  for (int pt = 0; pt < Tt; ++pt) {
    const float* krow = K + (((size_t)b * Tt + pt) * Nn + n) * Dd + ho;
    float acc = 0.f;
#pragma unroll
    for (int d = 0; d < 16; ++d) acc += qv[d] * krow[d];
    s[pt] = acc * 0.25f;                       // 1/sqrt(16)
    mx = fmaxf(mx, s[pt]);
  }
  float sum = 0.f;
  for (int pt = 0; pt < Tt; ++pt) { s[pt] = __expf(s[pt] - mx); sum += s[pt]; }
  float inv = 1.f / sum;
  float o[16];
#pragma unroll
  for (int d = 0; d < 16; ++d) o[d] = 0.f;
  for (int pt = 0; pt < Tt; ++pt) {
    const float* vrow = V + (((size_t)b * Tt + pt) * Nn + n) * Dd + ho;
    float w = s[pt] * inv;
#pragma unroll
    for (int d = 0; d < 16; ++d) o[d] += w * vrow[d];
  }
  float* orow = out + (((size_t)b * Tt + qt) * Nn + n) * Dd + ho;
#pragma unroll
  for (int d = 0; d < 16; ++d) orow[d] = o[d];
}

// ---------------------------------------------------------------------------
// Spatial attention over N=400 with graph bias. One wave per
// (b,t,head, 16-row tile). WMMA QK^T (d=16 zero-padded to K=32) -> LDS f32
// scores -> softmax -> f16 probs in LDS -> WMMA P*V over padded K=416 with
// pre-transposed V (contiguous B loads).
// ---------------------------------------------------------------------------
__global__ __launch_bounds__(32) void sattn_kernel(
    const float* __restrict__ Q, const float* __restrict__ K,
    const float* __restrict__ Vt, const float* __restrict__ biasNM,
    float* __restrict__ out)
{
  __shared__ float    sS[16 * NPAD];
  __shared__ _Float16 sA[16 * NPAD];
  int bid = blockIdx.x;
  int ntile = bid % 25; int tmp = bid / 25;
  int kk = tmp % Kh;    int bt  = tmp / Kh;       // bt in [0, B*T)
  int lane = threadIdx.x, r = lane & 15, h = lane >> 4;
  size_t tokbase = (size_t)bt * Nn;
  int n0 = ntile * 16, ho = kk * 16;

  // A = Q rows (d=16: K=h*8+j for elems 0..7 covers K0..15; upper half zero)
  v16h aQ;
  {
    const float* qrow = Q + (tokbase + n0 + r) * Dd + ho;
#pragma unroll
    for (int j = 0; j < 8; ++j) aQ[j] = (_Float16)qrow[h * 8 + j];
#pragma unroll
    for (int j = 0; j < 8; ++j) aQ[8 + j] = (_Float16)0.f;
  }
  for (int mt = 0; mt < 25; ++mt) {
    int m0 = mt * 16;
    v16h bK;
    const float* krow = K + (tokbase + m0 + r) * Dd + ho;
#pragma unroll
    for (int j = 0; j < 16; ++j) bK[j] = (j < 8) ? (_Float16)krow[h * 8 + j] : (_Float16)0.f;
    v8f c = v8f_zero();
    c = WMMA_F16(aQ, bK, c);
#pragma unroll
    for (int g = 0; g < 8; ++g) {
      int row = g + 8 * h, col = m0 + r;
      sS[row * NPAD + col] = c[g] * 0.25f + biasNM[(size_t)(n0 + row) * Nn + col];
    }
  }
  __syncthreads();
  // softmax each of the 16 rows (all 32 lanes cooperate per row)
  for (int row = 0; row < 16; ++row) {
    float mx = -1e30f;
    for (int cidx = lane; cidx < Nn; cidx += 32) mx = fmaxf(mx, sS[row * NPAD + cidx]);
#pragma unroll
    for (int off = 16; off > 0; off >>= 1) mx = fmaxf(mx, __shfl_xor(mx, off, 32));
    float sum = 0.f;
    for (int cidx = lane; cidx < Nn; cidx += 32) {
      float e = __expf(sS[row * NPAD + cidx] - mx);
      sS[row * NPAD + cidx] = e; sum += e;
    }
#pragma unroll
    for (int off = 16; off > 0; off >>= 1) sum += __shfl_xor(sum, off, 32);
    float inv = 1.f / sum;
    for (int cidx = lane; cidx < NPAD; cidx += 32)
      sA[row * NPAD + cidx] = (cidx < Nn) ? (_Float16)(sS[row * NPAD + cidx] * inv)
                                          : (_Float16)0.f;
  }
  __syncthreads();
  // out = P @ V  (B fragments contiguous in transposed V; pad cols are zero)
  const float* vtb = Vt + ((size_t)(bt * Kh + kk) * 16 + r) * NPAD;
  v8f acc = v8f_zero();
  for (int k0 = 0; k0 < NPAD; k0 += 32) {
    v16h a;
#pragma unroll
    for (int j = 0; j < 8; ++j) a[j]     = sA[r * NPAD + k0 + h * 8 + j];
#pragma unroll
    for (int j = 0; j < 8; ++j) a[8 + j] = sA[r * NPAD + k0 + 16 + h * 8 + j];
    v16h b;
#pragma unroll
    for (int j = 0; j < 8; ++j) b[j]     = (_Float16)vtb[k0 + h * 8 + j];
#pragma unroll
    for (int j = 0; j < 8; ++j) b[8 + j] = (_Float16)vtb[k0 + 16 + h * 8 + j];
    acc = WMMA_F16(a, b, acc);
  }
#pragma unroll
  for (int g = 0; g < 8; ++g)
    out[(tokbase + n0 + g + 8 * h) * Dd + ho + r] = acc[g];
}

// ---------------------------------------------------------------------------
// Host-side orchestration.
// Input index map (top-level dict insertion order; `params` tree-flattened
// with alphabetically sorted keys — note ASCII 'W' < 'b'):
//  0 x  1 te  2 lap_matrix  3 node_embedding
//  gf : 4 g1W 5 g1b 6 g2W 7 g2b 8 ln1b 9 ln1g 10 ln2b 11 ln2g 12 w1W 13 w1b 14 w2W 15 w2b
//  inc: 16 br0W 17 br0b 18 br1W 19 br1b 20 br2W 21 br2b 22 br3W 23 br3b
//       24 c0W 25 c0b 26 mW 27 pW 28 pb
//  sa : 29 alpha 30 beta 31 f1W 32 f1b 33 f2W 34 f2b 35 kW 36 kb 37 ln1b 38 ln1g
//       39 ln2b 40 ln2g 41 oW 42 ob 43 qW 44 qb 45 vW 46 vb
//  ta : 47 c0W 48 c0b 49 c1W 50 c1b 51 f1W 52 f1b 53 f2W 54 f2b 55 k4W 56 k4b
//       57 kW 58 kb 59 ln1b 60 ln1g 61 ln2b 62 ln2g 63 oW 64 ob 65 q4W 66 q4b
//       67 qW 68 qb 69 vW 70 vb
// Workspace: 9*M*128 + 400*400 + 24*128*128 + 768*16*416 floats ~= 200 MiB.
// ---------------------------------------------------------------------------
template<int KIN, int NOUT, bool RELU, bool HAS_RESID>
static inline void lin_t(const float* A, const float* W, const float* b,
                         const float* resid, float rs, float* out, hipStream_t s)
{
  constexpr int NT = NOUT / 64;
  gemm_lin_t<KIN, NOUT, RELU, HAS_RESID>
      <<<(Mtok / 16) * NT / 4, 128, 0, s>>>(A, W, b, resid, rs, out);
}
static inline void repack(const float* Win, float* Wout, int KK, hipStream_t s)
{
  repack_conv_kernel<<<(Dd * Dd * KK + 255) / 256, 256, 0, s>>>(Win, Wout, KK);
}

extern "C" void kernel_launch(void* const* d_in, const int* in_sizes, int n_in,
                              void* d_out, int out_size, void* d_ws, size_t ws_size,
                              hipStream_t stream)
{
  (void)in_sizes; (void)n_in; (void)out_size; (void)ws_size;
  const float* IN[71];
  for (int i = 0; i < 71; ++i) IN[i] = (const float*)d_in[i];
  const float* x   = IN[0];
  const float* te  = IN[1];
  const float* lap = IN[2];
  const float* ne  = IN[3];

  float* ws = (float*)d_ws;
  const size_t S = (size_t)Mtok * Dd;
  float* bufQ   = ws;
  float* bufQ2  = ws + 1 * S;
  float* bufK   = ws + 2 * S;
  float* bufK2  = ws + 3 * S;
  float* bufV   = ws + 4 * S;
  float* bufFFN = ws + 5 * S;         // M x 512 (also reused as M x 256 cat)
  float* bufB   = ws + 9 * S;         // 400 x 400 graph bias
  float* wpC0   = bufB + Nn * Nn;     // repacked conv weights
  float* wpC1   = wpC0 + 3 * Dd * Dd;
  float* wpB0   = wpC1 + 3 * Dd * Dd;
  float* wpB1   = wpB0 + 2 * Dd * Dd;
  float* wpB2   = wpB1 + 3 * Dd * Dd;
  float* wpB3   = wpB2 + 6 * Dd * Dd;
  float* bufVT  = wpB3 + 7 * Dd * Dd; // 768 x 16 x 416 transposed V

  const int EBLK = (Mtok * Dd + 255) / 256;
  const int CVB  = (Mtok / 16) * 2 / 4;         // conv blocks (16x64 tiles)
  const int LNB  = Mtok / 4;

  // repack all conv weights up front (independent of activations)
  repack(IN[47], wpC0, 3, stream);
  repack(IN[49], wpC1, 3, stream);
  repack(IN[16], wpB0, 2, stream);
  repack(IN[18], wpB1, 3, stream);
  repack(IN[20], wpB2, 6, stream);
  repack(IN[22], wpB3, 7, stream);

  // ================= temporal attention =================
  lin_t<Dd, Dd, false, false>(x, IN[67], IN[68], nullptr, 0.f, bufQ, stream);   // q
  add_te_kernel<<<EBLK, 256, 0, stream>>>(bufQ, te, IN[65], IN[66]);            // += te@q4W+q4b
  lin_t<Dd, Dd, false, false>(x, IN[57], IN[58], nullptr, 0.f, bufK, stream);   // k
  add_te_kernel<<<EBLK, 256, 0, stream>>>(bufK, te, IN[55], IN[56]);
  lin_t<Dd, Dd, false, false>(x, IN[69], IN[70], nullptr, 0.f, bufV, stream);   // v
  gemm_ctxconv<<<CVB, 128, 0, stream>>>(bufQ,  wpC0, IN[48], bufQ2);            // q: c0
  gemm_ctxconv<<<CVB, 128, 0, stream>>>(bufQ2, wpC1, IN[50], bufQ);             // q: c1
  gemm_ctxconv<<<CVB, 128, 0, stream>>>(bufK,  wpC0, IN[48], bufK2);            // k: c0
  gemm_ctxconv<<<CVB, 128, 0, stream>>>(bufK2, wpC1, IN[50], bufK);             // k: c1
  tattn_kernel<<<(Bb*Nn*Kh*Tt + 255) / 256, 256, 0, stream>>>(bufQ, bufK, bufV, bufQ2);
  lin_t<Dd, Dd, false, true>(bufQ2, IN[63], IN[64], x, 1.f, bufK2, stream);     // x1 = x+out@oW
  ln_t<false><<<LNB, 128, 0, stream>>>(bufK2, IN[60], IN[59], nullptr, bufQ);   // x1n
  lin_t<Dd, 512, true, false>(bufQ, IN[51], IN[52], nullptr, 0.f, bufFFN, stream); // relu(f1)
  lin_t<512, Dd, false, true>(bufFFN, IN[53], IN[54], bufQ, 1.f, bufK, stream); // x1n + f2
  ln_t<false><<<LNB, 128, 0, stream>>>(bufK, IN[62], IN[61], nullptr, bufV);    // ht1

  // ================= inception branch =================
  lin_t<Dd, Dd, false, false>(x, IN[24], IN[25], nullptr, 0.f, bufQ, stream);   // y0 (k=1 conv)
  merge_init_kernel<<<EBLK, 256, 0, stream>>>(bufK, bufQ, IN[26]);              // merged=mW0*y0
  gemm_causalconv<2><<<CVB, 128, 0, stream>>>(bufQ, wpB0, IN[17], IN[26] + 1, bufK);
  gemm_causalconv<3><<<CVB, 128, 0, stream>>>(bufQ, wpB1, IN[19], IN[26] + 2, bufK);
  gemm_causalconv<6><<<CVB, 128, 0, stream>>>(bufQ, wpB2, IN[21], IN[26] + 3, bufK);
  gemm_causalconv<7><<<CVB, 128, 0, stream>>>(bufQ, wpB3, IN[23], IN[26] + 4, bufK);
  lin_t<Dd, Dd, false, false>(bufK, IN[27], IN[28], nullptr, 0.f, bufQ2, stream); // ht2

  // ================= gated fusion =================
  ln_t<false><<<LNB, 128, 0, stream>>>(bufV, IN[9], IN[8], nullptr, bufQ);
  lin_t<Dd, Dd, false, true>(bufQ, IN[12], IN[13], bufV, 2.f, bufK, stream);    // h1
  ln_t<false><<<LNB, 128, 0, stream>>>(bufQ2, IN[11], IN[10], nullptr, bufQ);
  lin_t<Dd, Dd, false, true>(bufQ, IN[14], IN[15], bufQ2, 2.f, bufK2, stream);  // h2
  concat_kernel<<<EBLK, 256, 0, stream>>>(bufK, bufK2, bufFFN);                 // cat [M,256]
  lin_t<256, Dd, true, false>(bufFFN, IN[4], IN[5], nullptr, 0.f, bufQ, stream);// relu(g1)
  lin_t<Dd, Dd, false, false>(bufQ, IN[6], IN[7], nullptr, 0.f, bufV, stream);  // gate logits
  gate_kernel<<<EBLK, 256, 0, stream>>>(bufV, bufK, bufK2, bufQ2);              // ht

  // ================= spatial attention =================
  lin_t<Dd, Dd, false, false>(bufQ2, IN[43], IN[44], nullptr, 0.f, bufQ, stream); // sq
  lin_t<Dd, Dd, false, false>(bufQ2, IN[35], IN[36], nullptr, 0.f, bufK, stream); // sk
  lin_t<Dd, Dd, false, false>(bufQ2, IN[45], IN[46], nullptr, 0.f, bufV, stream); // sv
  graph_bias_kernel<<<(Nn*Nn + 255) / 256, 256, 0, stream>>>(ne, lap, IN[29], IN[30], bufB);
  vtrans_kernel<<<(Bb*Tt*Kh*16*NPAD + 255) / 256, 256, 0, stream>>>(bufV, bufVT);
  sattn_kernel<<<Bb * Tt * Kh * 25, 32, 0, stream>>>(bufQ, bufK, bufVT, bufB, bufK2);
  lin_t<Dd, Dd, false, true>(bufK2, IN[41], IN[42], bufQ2, 1.f, bufQ, stream);  // x1 = ht + o
  ln_t<false><<<LNB, 128, 0, stream>>>(bufQ, IN[38], IN[37], nullptr, bufK);    // x1n
  lin_t<Dd, 512, true, false>(bufK, IN[31], IN[32], nullptr, 0.f, bufFFN, stream); // relu(f1)
  lin_t<512, Dd, false, true>(bufFFN, IN[33], IN[34], bufK, 1.f, bufV, stream); // x1n + f2
  // final: out = x_raw + LN(hspre)
  ln_t<true><<<LNB, 128, 0, stream>>>(bufV, IN[40], IN[39], x, (float*)d_out);
}
